// SetTransformerEncoder_63926293234273
// MI455X (gfx1250) — compile-verified, matrix operand order checked by reading, not run on hardware
//
#include <hip/hip_runtime.h>
#include <hip/hip_bf16.h>

// Problem constants (match reference)
#define LAYERS 4
#define DIM    512
#define HEADS  8
#define DHEAD  64
#define OUTD   256
#define BATCH  8
#define SEQ    1024
#define ROWS   (BATCH * SEQ)   // 8192
#define LNEPS  1e-6f

typedef __attribute__((ext_vector_type(16))) __bf16 v16bf;
typedef __attribute__((ext_vector_type(8)))  float  v8f;

static __device__ __forceinline__ int lane_id() { return threadIdx.x & 31; }

static __device__ __forceinline__ v8f wmma_bf16(v16bf a, v16bf b, v8f c) {
  // (neg_a, A, neg_b, B, c_mod, C, reuse_a, reuse_b)
  return __builtin_amdgcn_wmma_f32_16x16x32_bf16(false, a, false, b, (short)0, c,
                                                 false, false);
}

// 16x32 bf16 A-fragment from row-major source (also serves as 32x16 B-fragment
// when the source is stored column-major: lane -> column, contiguous K).
// ISA layout: lane L: row = L&15, half = L>>4; VGPRs 0-3 hold K = 8*half+{0..7},
// VGPRs 4-7 hold K = 16+8*half+{0..7}  ->  two contiguous 16-byte chunks.
static __device__ __forceinline__ v16bf load_frag_bf16(const __bf16* __restrict__ src,
                                                       int ld) {
  const int lane = lane_id();
  const int r    = lane & 15;
  const int half = lane >> 4;
  const __bf16* row = src + (size_t)r * ld;
  v16bf f;
#pragma unroll
  for (int v = 0; v < 4; ++v) {
    const int k = 8 * half + 2 * v;
    f[2 * v    ] = row[k];
    f[2 * v + 1] = row[k + 1];
    f[2 * v + 8] = row[k + 16];
    f[2 * v + 9] = row[k + 17];
  }
  return f;
}

// Same fragment, but source is f32 (LDS softmax probs), converted on the fly.
static __device__ __forceinline__ v16bf load_frag_f32(const float* __restrict__ src,
                                                      int ld) {
  const int lane = lane_id();
  const int r    = lane & 15;
  const int half = lane >> 4;
  const float* row = src + (size_t)r * ld;
  v16bf f;
#pragma unroll
  for (int v = 0; v < 4; ++v) {
    const int k = 8 * half + 2 * v;
    f[2 * v    ] = (__bf16)row[k];
    f[2 * v + 1] = (__bf16)row[k + 1];
    f[2 * v + 8] = (__bf16)row[k + 16];
    f[2 * v + 9] = (__bf16)row[k + 17];
  }
  return f;
}

// ---------------------------------------------------------------------------
// Elementwise converts
// ---------------------------------------------------------------------------
__global__ void k_conv_bf16(const float* __restrict__ src, __bf16* __restrict__ dst,
                            long n) {
  long i = (long)blockIdx.x * blockDim.x + threadIdx.x;
  if (i < n) dst[i] = (__bf16)src[i];
}

// src: [layers][rows][cols] row-major f32  ->  dst: [layers][cols][rows] bf16
__global__ void k_conv_transpose_bf16(const float* __restrict__ src,
                                      __bf16* __restrict__ dst,
                                      int rows, int cols, int layers) {
  long per = (long)rows * cols;
  long i = (long)blockIdx.x * blockDim.x + threadIdx.x;
  if (i >= per * layers) return;
  int  l  = (int)(i / per);
  long rr = i % per;
  int  r  = (int)(rr / cols);
  int  c  = (int)(rr % cols);
  dst[l * per + (long)c * rows + r] = (__bf16)src[i];
}

// ---------------------------------------------------------------------------
// GEMM: out = epilogue(A[M][K]_bf16 @ W[Ncols][K]_bf16(col-major) + bias)
// Wave tile: 32 rows x 64 cols (2 A-frags x 4 B-frags = 8 wmma / k-step).
// Block = 4 waves = 128x64 tile.
// ---------------------------------------------------------------------------
__global__ void k_gemm_wmma(const __bf16* __restrict__ A,
                            const __bf16* __restrict__ Wt,
                            const float*  __restrict__ bias,
                            const float*  __restrict__ resid,   // may be null
                            float*        __restrict__ outF,    // may be null
                            __bf16*       __restrict__ outB,    // may be null
                            __bf16*       __restrict__ outBT,   // [B][Ncols][trRows], may be null
                            int M, int Ncols, int K, int trRows, int doRelu) {
  const int wave = threadIdx.x >> 5;
  const int row0 = blockIdx.x * 128 + wave * 32;
  const int col0 = blockIdx.y * 64;
  if (row0 >= M) return;

  v8f acc[2][4];
#pragma unroll
  for (int mi = 0; mi < 2; ++mi)
#pragma unroll
    for (int t = 0; t < 4; ++t)
#pragma unroll
      for (int i = 0; i < 8; ++i) acc[mi][t][i] = 0.f;

  for (int k0 = 0; k0 < K; k0 += 32) {
    v16bf a0 = load_frag_bf16(A + (size_t)row0 * K + k0, K);
    v16bf a1 = load_frag_bf16(A + (size_t)(row0 + 16) * K + k0, K);
#pragma unroll
    for (int t = 0; t < 4; ++t) {
      v16bf bf = load_frag_bf16(Wt + (size_t)(col0 + 16 * t) * K + k0, K);
      acc[0][t] = wmma_bf16(a0, bf, acc[0][t]);
      acc[1][t] = wmma_bf16(a1, bf, acc[1][t]);
    }
  }

  const int lane = lane_id();
  const int nn   = lane & 15;
  const int half = lane >> 4;
#pragma unroll
  for (int mi = 0; mi < 2; ++mi) {
#pragma unroll
    for (int t = 0; t < 4; ++t) {
      const int col = col0 + 16 * t + nn;
#pragma unroll
      for (int r = 0; r < 8; ++r) {
        const int m = row0 + 16 * mi + 8 * half + r;
        float v = acc[mi][t][r] + bias[col];
        if (doRelu) v = fmaxf(v, 0.f);
        if (resid)  v += resid[(size_t)m * Ncols + col];
        const size_t idx = (size_t)m * Ncols + col;
        if (outF) outF[idx] = v;
        if (outB) outB[idx] = (__bf16)v;
        if (outBT) {
          const int bb = m / trRows;
          const int nr = m - bb * trRows;
          outBT[((size_t)bb * Ncols + col) * trRows + nr] = (__bf16)v;
        }
      }
    }
  }
}

// ---------------------------------------------------------------------------
// Attention: per (b, h, 16-row tile). Scores in LDS (16x1024 f32 = 64 KB),
// exact mask penalty, full softmax, P@V via wmma, residual o = q + attn.
// Block = 256 threads = 8 waves. Phase 3 uses all 8 waves (d-slice x K-half),
// reducing K-half partials through the (already consumed) score buffer.
// ---------------------------------------------------------------------------
__global__ void k_attention_wmma(const __bf16* __restrict__ Qb,  // [B][N][D]
                                 const __bf16* __restrict__ Kb,  // [B][N][D]
                                 const __bf16* __restrict__ Vt,  // [B][D][N]
                                 const float*  __restrict__ Qf,  // [B][N][D]
                                 const int*    __restrict__ mask,// [B][N]
                                 float*        __restrict__ O) { // [B][N][D]
  __shared__ float sS[16 * SEQ];   // 64 KB

  const int r0   = blockIdx.x * 16;
  const int h    = blockIdx.y;
  const int b    = blockIdx.z;
  const int wave = threadIdx.x >> 5;
  const int lane = lane_id();
  const int nn   = lane & 15;
  const int half = lane >> 4;
  const float scale = 0.04419417382415922f;   // 1/sqrt(512) (reference scales by sqrt(D))

  const __bf16* qbase = Qb + ((size_t)(b * SEQ + r0)) * DIM + h * DHEAD;
  const v16bf aq0 = load_frag_bf16(qbase,      DIM);
  const v16bf aq1 = load_frag_bf16(qbase + 32, DIM);

  int mq[8];
#pragma unroll
  for (int r = 0; r < 8; ++r) mq[r] = mask[b * SEQ + r0 + 8 * half + r];

  // ---- phase 1: S = Q K^T * scale + pen  (each wave does 8 of 64 col tiles)
  for (int ct = wave; ct < SEQ / 16; ct += 8) {
    const int n0 = ct * 16;
    const __bf16* kbase = Kb + ((size_t)(b * SEQ + n0)) * DIM + h * DHEAD;
    v16bf bk0 = load_frag_bf16(kbase,      DIM);
    v16bf bk1 = load_frag_bf16(kbase + 32, DIM);
    v8f acc;
#pragma unroll
    for (int i = 0; i < 8; ++i) acc[i] = 0.f;
    acc = wmma_bf16(aq0, bk0, acc);
    acc = wmma_bf16(aq1, bk1, acc);

    const int kcol = n0 + nn;
    const int mk = mask[b * SEQ + kcol];
#pragma unroll
    for (int r = 0; r < 8; ++r) {
      const int m = 8 * half + r;
      float s = acc[r] * scale;
      if (mq[r] && mk) s -= 1e9f;      // pen = -1e9 * m_q * m_k
      sS[m * SEQ + kcol] = s;
    }
  }
  __syncthreads();

  // ---- phase 2: softmax per row (2 rows per wave), normalize in place
  for (int rr = 0; rr < 2; ++rr) {
    float* srow = sS + (wave * 2 + rr) * SEQ;
    float mx = -3.4e38f;
    for (int i = lane; i < SEQ; i += 32) mx = fmaxf(mx, srow[i]);
    for (int off = 16; off > 0; off >>= 1) mx = fmaxf(mx, __shfl_xor(mx, off, 32));
    float sum = 0.f;
    for (int i = lane; i < SEQ; i += 32) {
      float e = __expf(srow[i] - mx);
      srow[i] = e;
      sum += e;
    }
    for (int off = 16; off > 0; off >>= 1) sum += __shfl_xor(sum, off, 32);
    const float inv = 1.f / sum;
    for (int i = lane; i < SEQ; i += 32) srow[i] *= inv;
  }
  __syncthreads();

  // ---- phase 3: O = Q + P @ V.  wave -> (d-slice, K-half)
  const int d0 = (wave & 3) * 16;
  const int kh = wave >> 2;
  v8f acc;
#pragma unroll
  for (int i = 0; i < 8; ++i) acc[i] = 0.f;
  const __bf16* vbase = Vt + ((size_t)b * DIM + h * DHEAD + d0) * SEQ;
  const int kbeg = kh * (SEQ / 2);
  for (int k0 = kbeg; k0 < kbeg + SEQ / 2; k0 += 32) {
    v16bf ap = load_frag_f32(sS + k0, SEQ);
    v16bf bv = load_frag_bf16(vbase + k0, SEQ);
    acc = wmma_bf16(ap, bv, acc);
  }
  __syncthreads();   // all probs consumed; first 4 KB of sS reusable as partials

  float* part = sS;  // [16][DHEAD] f32 partial sums (1024 floats)
  if (kh == 1) {
#pragma unroll
    for (int r = 0; r < 8; ++r)
      part[(8 * half + r) * DHEAD + d0 + nn] = acc[r];
  }
  __syncthreads();
  if (kh == 0) {
#pragma unroll
    for (int r = 0; r < 8; ++r) {
      const int m = 8 * half + r;
      const size_t idx =
          ((size_t)(b * SEQ + r0 + m)) * DIM + h * DHEAD + d0 + nn;
      O[idx] = Qf[idx] + acc[r] + part[m * DHEAD + d0 + nn];
    }
  }
}

// ---------------------------------------------------------------------------
// LayerNorm: one wave per row of 512; writes f32 + bf16.
// Block = 256 threads = 8 rows.
// ---------------------------------------------------------------------------
__global__ void k_layernorm(const float* __restrict__ src,
                            const float* __restrict__ g,
                            const float* __restrict__ beta,
                            float*       __restrict__ dstF,
                            __bf16*      __restrict__ dstB) {
  const int row  = blockIdx.x * 8 + (threadIdx.x >> 5);
  const int lane = lane_id();
  const float* x = src + (size_t)row * DIM + lane * 16;

  float xv[16];
  float s = 0.f;
#pragma unroll
  for (int j = 0; j < 16; ++j) { xv[j] = x[j]; s += xv[j]; }
  for (int off = 16; off > 0; off >>= 1) s += __shfl_xor(s, off, 32);
  const float mean = s * (1.f / DIM);

  float vs = 0.f;
#pragma unroll
  for (int j = 0; j < 16; ++j) { float d = xv[j] - mean; vs += d * d; }
  for (int off = 16; off > 0; off >>= 1) vs += __shfl_xor(vs, off, 32);
  const float rs = rsqrtf(vs * (1.f / DIM) + LNEPS);

#pragma unroll
  for (int j = 0; j < 16; ++j) {
    const int c = lane * 16 + j;
    const float y = (xv[j] - mean) * rs * g[c] + beta[c];
    dstF[(size_t)row * DIM + c] = y;
    dstB[(size_t)row * DIM + c] = (__bf16)y;
  }
}

// ---------------------------------------------------------------------------
// Host orchestration
// ---------------------------------------------------------------------------
extern "C" void kernel_launch(void* const* d_in, const int* in_sizes, int n_in,
                              void* d_out, int out_size, void* d_ws, size_t ws_size,
                              hipStream_t stream) {
  (void)in_sizes; (void)n_in; (void)out_size; (void)ws_size;

  const float* inputs = (const float*)d_in[0];
  const int*   mask   = (const int*)d_in[1];
  const float* Wq   = (const float*)d_in[2];
  const float* bq   = (const float*)d_in[3];
  const float* Wk   = (const float*)d_in[4];
  const float* bk   = (const float*)d_in[5];
  const float* Wv   = (const float*)d_in[6];
  const float* bv   = (const float*)d_in[7];
  const float* Wo   = (const float*)d_in[8];
  const float* bo   = (const float*)d_in[9];
  const float* g0   = (const float*)d_in[10];
  const float* be0  = (const float*)d_in[11];
  const float* g1   = (const float*)d_in[12];
  const float* be1  = (const float*)d_in[13];
  const float* Wout = (const float*)d_in[14];
  const float* bout = (const float*)d_in[15];

  const size_t E = (size_t)ROWS * DIM;   // 4,194,304 elements

  char* ws = (char*)d_ws;
  size_t off = 0;
  auto take = [&](size_t bytes) -> void* {
    void* p = ws + off;
    off += (bytes + 255) & ~(size_t)255;
    return p;
  };
  float*  xF    = (float*) take(E * 4);                          // LN0 output H (f32)
  float*  qF    = (float*) take(E * 4);                          // projected q (f32)
  float*  oF    = (float*) take(E * 4);                          // attn / ffn output (f32)
  __bf16* xB    = (__bf16*)take(E * 2);                          // activations bf16
  __bf16* qB    = (__bf16*)take(E * 2);
  __bf16* kB    = (__bf16*)take(E * 2);
  __bf16* vT    = (__bf16*)take(E * 2);                          // [B][D][N]
  __bf16* WqB   = (__bf16*)take((size_t)LAYERS * DIM * DIM * 2); // col-major per layer
  __bf16* WkB   = (__bf16*)take((size_t)LAYERS * DIM * DIM * 2);
  __bf16* WvB   = (__bf16*)take((size_t)LAYERS * DIM * DIM * 2);
  __bf16* WoB   = (__bf16*)take((size_t)LAYERS * DIM * DIM * 2);
  __bf16* WoutB = (__bf16*)take((size_t)DIM * OUTD * 2);

  const int TP = 256;
  const long WN = (long)LAYERS * DIM * DIM;
  k_conv_transpose_bf16<<<(int)((WN + TP - 1) / TP), TP, 0, stream>>>(Wq, WqB, DIM, DIM, LAYERS);
  k_conv_transpose_bf16<<<(int)((WN + TP - 1) / TP), TP, 0, stream>>>(Wk, WkB, DIM, DIM, LAYERS);
  k_conv_transpose_bf16<<<(int)((WN + TP - 1) / TP), TP, 0, stream>>>(Wv, WvB, DIM, DIM, LAYERS);
  k_conv_transpose_bf16<<<(int)((WN + TP - 1) / TP), TP, 0, stream>>>(Wo, WoB, DIM, DIM, LAYERS);
  k_conv_transpose_bf16<<<(DIM * OUTD + TP - 1) / TP, TP, 0, stream>>>(Wout, WoutB, DIM, OUTD, 1);
  k_conv_bf16<<<(int)((E + TP - 1) / TP), TP, 0, stream>>>(inputs, xB, (long)E);

  dim3 gGemm(ROWS / 128, DIM / 64);
  dim3 gAttn(SEQ / 16, HEADS, BATCH);
  dim3 gLN(ROWS / 8);

  for (int l = 0; l < LAYERS; ++l) {
    const __bf16* wq = WqB + (size_t)l * DIM * DIM;
    const __bf16* wk = WkB + (size_t)l * DIM * DIM;
    const __bf16* wv = WvB + (size_t)l * DIM * DIM;
    const __bf16* wo = WoB + (size_t)l * DIM * DIM;

    // q = x@Wq + bq  (f32 + bf16)
    k_gemm_wmma<<<gGemm, 128, 0, stream>>>(xB, wq, bq + l * DIM, nullptr,
                                           qF, qB, nullptr, ROWS, DIM, DIM, SEQ, 0);
    // k = x@Wk + bk  (bf16)
    k_gemm_wmma<<<gGemm, 128, 0, stream>>>(xB, wk, bk + l * DIM, nullptr,
                                           nullptr, kB, nullptr, ROWS, DIM, DIM, SEQ, 0);
    // v = x@Wv + bv  (bf16, transposed [B][D][N])
    k_gemm_wmma<<<gGemm, 128, 0, stream>>>(xB, wv, bv + l * DIM, nullptr,
                                           nullptr, nullptr, vT, ROWS, DIM, DIM, SEQ, 0);
    // o = q + softmax(qk^T*scale + pen) v
    k_attention_wmma<<<gAttn, 256, 0, stream>>>(qB, kB, vT, qF, mask, oF);
    // H = LN(o; g0, beta0)
    k_layernorm<<<gLN, 256, 0, stream>>>(oF, g0 + l * DIM, be0 + l * DIM, xF, xB);
    // o2 = H + relu(H@Wo + bo)
    k_gemm_wmma<<<gGemm, 128, 0, stream>>>(xB, wo, bo + l * DIM, xF,
                                           oF, nullptr, nullptr, ROWS, DIM, DIM, SEQ, 1);
    // x = LN(o2; g1, beta1)
    k_layernorm<<<gLN, 256, 0, stream>>>(oF, g1 + l * DIM, be1 + l * DIM, xF, xB);
  }

  // out = x @ Wout + bout
  dim3 gOut(ROWS / 128, OUTD / 64);
  k_gemm_wmma<<<gOut, 128, 0, stream>>>(xB, WoutB, bout, nullptr,
                                        (float*)d_out, nullptr, nullptr,
                                        ROWS, OUTD, DIM, SEQ, 0);
}
